// I64MLP_41420664603008
// MI455X (gfx1250) — matrix operand order repaired and checked by reading, not compiled, not run on hardware
//
#include <hip/hip_runtime.h>

// ---------------------------------------------------------------------------
// int8-quantized MLP (gate/up/down) on gfx1250.
// GEMMs via V_WMMA_F32_16X16X32_F16; global->LDS staging via CDNA5 async
// copies (GLOBAL_LOAD_ASYNC_TO_LDS_B128, ASYNCcnt) with double buffering.
// Slot strides are carried in the instruction IOFFSET, which the ISA applies
// to BOTH the LDS and global addresses
// -> one address add per stream per step.
// ---------------------------------------------------------------------------

typedef _Float16 half_t;
typedef __attribute__((ext_vector_type(16))) _Float16 v16h;
typedef __attribute__((ext_vector_type(8)))  float    v8f;

union Frag { v16h h; uint4 u[2]; };

constexpr int kH = 2048;   // hidden
constexpr int kI = 8192;   // intermediate
constexpr int kT = 4096;   // tokens

constexpr int BM  = 128;   // M tile
constexpr int BK  = 64;    // K tile (halfs)
constexpr int BN1 = 64;    // N tile for gemm1 (per gate AND per up)
constexpr int BN2 = 128;   // N tile for gemm2
constexpr int LDT = BK + 8; // padded LDS row stride (halfs), keeps 16B align

// ---- CDNA5 async global->LDS copy (ASYNCcnt path) -------------------------
template <int OFF>
__device__ __forceinline__ void async_ld128(unsigned lds_addr, unsigned long long g) {
  asm volatile("global_load_async_to_lds_b128 %0, %1, off offset:%2"
               :: "v"(lds_addr), "v"(g), "n"(OFF) : "memory");
}
template <int N>
__device__ __forceinline__ void wait_asynccnt() {
  asm volatile("s_wait_asynccnt %0" :: "n"(N) : "memory");
}
__device__ __forceinline__ unsigned lds_off(const void* p) {
  // flat->LDS aperture mapping truncates to addr[31:0] (ISA 10.2)
  return (unsigned)(unsigned long long)p;
}

__device__ __forceinline__ float silu8(float g) {
  // reference LUT clamps input to [-8,8]; LUT-lerp error ~1e-6 vs exact silu
  g = fminf(fmaxf(g, -8.0f), 8.0f);
  return g * (1.0f / (1.0f + __expf(-g)));
}

// ---------------------------------------------------------------------------
// fp32 -> f16 conversion, 8 elements/thread, b128 store
// ---------------------------------------------------------------------------
__global__ __launch_bounds__(256) void cvt_f32_f16_kernel(
    const float* __restrict__ in, half_t* __restrict__ out, size_t n) {
  size_t i = ((size_t)blockIdx.x * blockDim.x + threadIdx.x) * 8;
  if (i + 8 > n) return;
  const float4* p = (const float4*)(in + i);
  float4 f0 = p[0], f1 = p[1];
  union { half_t h[8]; uint4 u; } r;
  r.h[0] = (half_t)f0.x; r.h[1] = (half_t)f0.y;
  r.h[2] = (half_t)f0.z; r.h[3] = (half_t)f0.w;
  r.h[4] = (half_t)f1.x; r.h[5] = (half_t)f1.y;
  r.h[6] = (half_t)f1.z; r.h[7] = (half_t)f1.w;
  *(uint4*)(out + i) = r.u;
}

// ---------------------------------------------------------------------------
// GEMM1 + fused dequant-scale + SiLU(gate)*up epilogue.
// C tile: 128 rows x 64 gate channels (and the matching 64 up channels).
// Double-buffered LDS, async staging; 8 waves, each 32M x 32N for gate+up.
// ---------------------------------------------------------------------------
__global__ __launch_bounds__(256) void gemm1_silu_kernel(
    const half_t* __restrict__ xh,    // T x H
    const half_t* __restrict__ wgu,   // 2I x H (row-major, channel rows)
    const float*  __restrict__ gus,   // 2I scales
    half_t* __restrict__ inter)       // T x I (f16 out)
{
  __shared__ half_t As[2][BM  * LDT];
  __shared__ half_t Bg[2][BN1 * LDT];
  __shared__ half_t Bu[2][BN1 * LDT];

  const int tid  = threadIdx.x;
  const int wave = tid >> 5;
  const int lane = tid & 31;
  const int m0 = blockIdx.x * BM;
  const int n0 = blockIdx.y * BN1;

  const int wm = (wave >> 1) * 32;
  const int wn = (wave & 1)  * 32;

  v8f accG[2][2] = {};
  v8f accU[2][2] = {};

  // cooperative staging slices: A 4x16B, Bg 2x16B, Bu 2x16B per thread
  const int arow = tid >> 1;           // 0..127, 32 halfs
  const int acol = (tid & 1) * 32;
  const int brow = tid >> 2;           // 0..63, 16 halfs
  const int bcol = (tid & 3) * 16;

  const half_t* aptr = xh  + (size_t)(m0 + arow) * kH + acol;
  const half_t* gptr = wgu + (size_t)(n0 + brow) * kH + bcol;
  const half_t* uptr = wgu + ((size_t)kI + n0 + brow) * kH + bcol;

  unsigned asl[2], bgl[2], bul[2];
#pragma unroll
  for (int b = 0; b < 2; ++b) {
    asl[b] = lds_off(&As[b][arow * LDT + acol]);
    bgl[b] = lds_off(&Bg[b][brow * LDT + bcol]);
    bul[b] = lds_off(&Bu[b][brow * LDT + bcol]);
  }

  auto issue = [&](int buf, int k0) {
    unsigned long long ga = (unsigned long long)(aptr + k0);
    unsigned long long gg = (unsigned long long)(gptr + k0);
    unsigned long long gu = (unsigned long long)(uptr + k0);
    async_ld128<0> (asl[buf], ga);
    async_ld128<16>(asl[buf], ga);
    async_ld128<32>(asl[buf], ga);
    async_ld128<48>(asl[buf], ga);
    async_ld128<0> (bgl[buf], gg);
    async_ld128<16>(bgl[buf], gg);
    async_ld128<0> (bul[buf], gu);
    async_ld128<16>(bul[buf], gu);
  };

  const int lm  = lane & 15;   // matrix row (A) / col (B,C)
  const int sel = lane >> 4;   // lane-half select

  auto compute = [&](const half_t* A_, const half_t* G_, const half_t* U_) {
#pragma unroll
    for (int kk = 0; kk < BK; kk += 32) {
      Frag a[2];
#pragma unroll
      for (int i = 0; i < 2; ++i) {
        const half_t* p = &A_[(wm + i * 16 + lm) * LDT + kk];
        a[i].u[0] = *(const uint4*)(p + sel * 16);       // K = sel*16 .. +8
        a[i].u[1] = *(const uint4*)(p + sel * 16 + 8);   // K = sel*16+8 .. +8
      }
      Frag bg[2], bu[2];
#pragma unroll
      for (int j = 0; j < 2; ++j) {
        const half_t* pg = &G_[(wn + j * 16 + lm) * LDT + kk + sel * 16];
        bg[j].u[0] = ((const uint4*)pg)[0];
        bg[j].u[1] = ((const uint4*)pg)[1];
        const half_t* pu = &U_[(wn + j * 16 + lm) * LDT + kk + sel * 16];
        bu[j].u[0] = ((const uint4*)pu)[0];
        bu[j].u[1] = ((const uint4*)pu)[1];
      }
#pragma unroll
      for (int i = 0; i < 2; ++i)
#pragma unroll
        for (int j = 0; j < 2; ++j) {
          accG[i][j] = __builtin_amdgcn_wmma_f32_16x16x32_f16(
              false, a[i].h, false, bg[j].h, (short)0, accG[i][j], false, false);
          accU[i][j] = __builtin_amdgcn_wmma_f32_16x16x32_f16(
              false, a[i].h, false, bu[j].h, (short)0, accU[i][j], false, false);
        }
    }
  };

  auto step = [&](int buf, int k0) {
    if (k0 + BK < kH) { issue(buf ^ 1, k0 + BK); wait_asynccnt<8>(); }
    else              { wait_asynccnt<0>(); }
    __syncthreads();
    compute(&As[buf][0], &Bg[buf][0], &Bu[buf][0]);
    __syncthreads();
  };

  issue(0, 0);
  for (int k0 = 0; k0 < kH; k0 += 2 * BK) {
    step(0, k0);
    step(1, k0 + BK);
  }

  // epilogue: scale, silu(gate)*up, store f16 inter
  const int rb = sel * 8;
#pragma unroll
  for (int j = 0; j < 2; ++j) {
    const int n = n0 + wn + j * 16 + lm;
    const float sg = gus[n];
    const float su = gus[kI + n];
#pragma unroll
    for (int i = 0; i < 2; ++i) {
#pragma unroll
      for (int v = 0; v < 8; ++v) {
        const int m = m0 + wm + i * 16 + rb + v;
        float g = accG[i][j][v] * sg;
        float u = accU[i][j][v] * su;
        inter[(size_t)m * kI + n] = (half_t)(silu8(g) * u);
      }
    }
  }
}

// ---------------------------------------------------------------------------
// GEMM2: out[T,H] = inter[T,I](f16) @ down_w[H,I]^T * down_scale, fp32 out.
// 128x128 tile, double-buffered async staging; 8 waves each 32M x 64N.
// ---------------------------------------------------------------------------
__global__ __launch_bounds__(256) void gemm2_kernel(
    const half_t* __restrict__ inter, // T x I
    const half_t* __restrict__ dwh,   // H x I
    const float*  __restrict__ dsc,   // H scales
    float* __restrict__ out)          // T x H
{
  __shared__ half_t As[2][BM  * LDT];
  __shared__ half_t Bs[2][BN2 * LDT];

  const int tid  = threadIdx.x;
  const int wave = tid >> 5;
  const int lane = tid & 31;
  const int m0 = blockIdx.x * BM;
  const int n0 = blockIdx.y * BN2;

  const int wm = (wave >> 1) * 32;
  const int wn = (wave & 1)  * 64;

  v8f acc[2][4] = {};

  const int row = tid >> 1;            // 0..127
  const int col = (tid & 1) * 32;      // 32 halfs (4 x b128) per thread
  const half_t* aptr = inter + (size_t)(m0 + row) * kI + col;
  const half_t* bptr = dwh   + (size_t)(n0 + row) * kI + col;

  unsigned asl[2], bsl[2];
#pragma unroll
  for (int b = 0; b < 2; ++b) {
    asl[b] = lds_off(&As[b][row * LDT + col]);
    bsl[b] = lds_off(&Bs[b][row * LDT + col]);
  }

  auto issue = [&](int buf, int k0) {
    unsigned long long ga = (unsigned long long)(aptr + k0);
    unsigned long long gb = (unsigned long long)(bptr + k0);
    async_ld128<0> (asl[buf], ga);
    async_ld128<16>(asl[buf], ga);
    async_ld128<32>(asl[buf], ga);
    async_ld128<48>(asl[buf], ga);
    async_ld128<0> (bsl[buf], gb);
    async_ld128<16>(bsl[buf], gb);
    async_ld128<32>(bsl[buf], gb);
    async_ld128<48>(bsl[buf], gb);
  };

  const int lm  = lane & 15;
  const int sel = lane >> 4;

  auto compute = [&](const half_t* A_, const half_t* B_) {
#pragma unroll
    for (int kk = 0; kk < BK; kk += 32) {
      Frag a[2];
#pragma unroll
      for (int i = 0; i < 2; ++i) {
        const half_t* p = &A_[(wm + i * 16 + lm) * LDT + kk];
        a[i].u[0] = *(const uint4*)(p + sel * 16);
        a[i].u[1] = *(const uint4*)(p + sel * 16 + 8);
      }
      Frag b[4];
#pragma unroll
      for (int j = 0; j < 4; ++j) {
        const half_t* pb = &B_[(wn + j * 16 + lm) * LDT + kk + sel * 16];
        b[j].u[0] = ((const uint4*)pb)[0];
        b[j].u[1] = ((const uint4*)pb)[1];
      }
#pragma unroll
      for (int i = 0; i < 2; ++i)
#pragma unroll
        for (int j = 0; j < 4; ++j)
          acc[i][j] = __builtin_amdgcn_wmma_f32_16x16x32_f16(
              false, a[i].h, false, b[j].h, (short)0, acc[i][j], false, false);
    }
  };

  auto step = [&](int buf, int k0) {
    if (k0 + BK < kI) { issue(buf ^ 1, k0 + BK); wait_asynccnt<8>(); }
    else              { wait_asynccnt<0>(); }
    __syncthreads();
    compute(&As[buf][0], &Bs[buf][0]);
    __syncthreads();
  };

  issue(0, 0);
  for (int k0 = 0; k0 < kI; k0 += 2 * BK) {
    step(0, k0);
    step(1, k0 + BK);
  }

  const int rb = sel * 8;
#pragma unroll
  for (int j = 0; j < 4; ++j) {
    const int n = n0 + wn + j * 16 + lm;
    const float s = dsc[n];
#pragma unroll
    for (int i = 0; i < 2; ++i) {
#pragma unroll
      for (int v = 0; v < 8; ++v) {
        const int m = m0 + wm + i * 16 + rb + v;
        out[(size_t)m * kH + n] = acc[i][j][v] * s;
      }
    }
  }
}

// ---------------------------------------------------------------------------
extern "C" void kernel_launch(void* const* d_in, const int* in_sizes, int n_in,
                              void* d_out, int out_size, void* d_ws, size_t ws_size,
                              hipStream_t stream) {
  const float* x    = (const float*)d_in[0];  // T x H
  const float* wgu  = (const float*)d_in[1];  // 2I x H
  const float* gus  = (const float*)d_in[2];  // 2I
  const float* dw   = (const float*)d_in[3];  // H x I
  const float* dsc  = (const float*)d_in[4];  // H
  float* out = (float*)d_out;

  half_t* xh    = (half_t*)d_ws;                       // T*H      = 8M  halfs
  half_t* wguh  = xh   + (size_t)kT * kH;              // 2I*H     = 32M halfs
  half_t* dwh   = wguh + (size_t)2 * kI * kH;          // H*I      = 16M halfs
  half_t* inter = dwh  + (size_t)kH * kI;              // T*I      = 32M halfs

  const size_t nx = (size_t)kT * kH;
  const size_t ng = (size_t)2 * kI * kH;
  const size_t nd = (size_t)kH * kI;
  cvt_f32_f16_kernel<<<(nx / 8 + 255) / 256, 256, 0, stream>>>(x,   xh,   nx);
  cvt_f32_f16_kernel<<<(ng / 8 + 255) / 256, 256, 0, stream>>>(wgu, wguh, ng);
  cvt_f32_f16_kernel<<<(nd / 8 + 255) / 256, 256, 0, stream>>>(dw,  dwh,  nd);

  dim3 g1(kT / BM, kI / BN1);   // 32 x 128
  gemm1_silu_kernel<<<g1, 256, 0, stream>>>(xh, wguh, gus, inter);

  dim3 g2(kT / BM, kH / BN2);   // 32 x 16
  gemm2_kernel<<<g2, 256, 0, stream>>>(inter, dwh, dsc, out);
}
